// Attention_51900384805169
// MI455X (gfx1250) — matrix operand order repaired and checked by reading, not compiled, not run on hardware
//
#include <hip/hip_runtime.h>

#define B_      8
#define C_      256
#define HGRID_  48
#define N_      2304
#define NH_     4
#define DK_     32
#define DV_     64
#define HQKV_   512
#define BN_EPS_ 0.001f
#define SCALE_  0.17677669529663687f   /* 32^-0.5 */
#define LDA_    40                     /* LDS row stride in bf16: 80B = 16B-aligned, conflict-free */

typedef __attribute__((ext_vector_type(16))) __bf16 v16bf;
typedef __attribute__((ext_vector_type(8)))  __bf16 v8bf;
typedef __attribute__((ext_vector_type(8)))  float  v8f;

__device__ __forceinline__ v8f wmma_bf16(v16bf a, v16bf b, v8f c) {
  return __builtin_amdgcn_wmma_f32_16x16x32_bf16(false, a, false, b, (short)0, c,
                                                 false, false);
}

__device__ __forceinline__ v16bf mk16(v8bf a, v8bf b) {
  return __builtin_shufflevector(a, b, 0, 1, 2, 3, 4, 5, 6, 7, 8, 9, 10, 11, 12, 13,
                                 14, 15);
}

// A fragment: logical a[m][k] (16x32) from row-major LDS tile (stride LDA_).
// lane L: m=L%16; elements are two contiguous 8-runs at k = hi*8 and 16+hi*8.
__device__ __forceinline__ v16bf load_a_frag(const __bf16* base, int lane) {
  const int m  = lane & 15;
  const int hi = (lane >> 4) & 1;
  const __bf16* r = base + m * LDA_ + hi * 8;
  return mk16(*(const v8bf*)r, *(const v8bf*)(r + 16));
}

// B fragment: logical b[k][n] (32x16) sourced from TRANSPOSED LDS tile [n][k]
// (stride LDA_). lane L: n = n0 + L%16; elements are two contiguous 8-runs at k = hi*8 and 16+hi*8.
__device__ __forceinline__ v16bf load_b_frag(const __bf16* base, int lane, int n0) {
  const int n  = n0 + (lane & 15);
  const int k0 = (lane >= 16) ? 8 : 0;
  const __bf16* r = base + n * LDA_ + k0;
  return mk16(*(const v8bf*)r, *(const v8bf*)(r + 16));
}

__device__ __forceinline__ float red_max16(float x) {
  x = fmaxf(x, __shfl_xor(x, 1, 32));
  x = fmaxf(x, __shfl_xor(x, 2, 32));
  x = fmaxf(x, __shfl_xor(x, 4, 32));
  x = fmaxf(x, __shfl_xor(x, 8, 32));
  return x;
}
__device__ __forceinline__ float red_sum16(float x) {
  x += __shfl_xor(x, 1, 32);
  x += __shfl_xor(x, 2, 32);
  x += __shfl_xor(x, 4, 32);
  x += __shfl_xor(x, 8, 32);
  return x;
}

// ---------------------------------------------------------------------------
// Kernel 1: QKV GEMM (bf16 WMMA) + folded BN. Block tile: 64 outs x 64 pixels,
// 8 waves = 4(o) x 2(n), 2 WMMAs per wave per k-step.
// Outputs: qT[bh][n][32], kT[bh][n][32], vW[bh][d][n] (bf16), vimg[b][c][n] f32.
// ---------------------------------------------------------------------------
__global__ __launch_bounds__(256) void qkv_gemm_kernel(
    const float* __restrict__ x, const float* __restrict__ w,
    const float* __restrict__ g, const float* __restrict__ bb,
    const float* __restrict__ mm, const float* __restrict__ vv,
    __bf16* __restrict__ qT, __bf16* __restrict__ kT,
    __bf16* __restrict__ vW, float* __restrict__ vimg) {
  __shared__ __align__(16) __bf16 Ws[64 * LDA_];   // [o][k] row-major
  __shared__ __align__(16) __bf16 Xs[64 * LDA_];   // [n][k] transposed
  __shared__ float inv_s[64];
  __shared__ float beta_s[64];

  const int b    = blockIdx.x;
  const int o0   = blockIdx.y * 64;
  const int n0   = blockIdx.z * 64;
  const int tid  = threadIdx.x;
  const int lane = tid & 31;
  const int wv   = tid >> 5;
  const int osub = (wv & 3) * 16;
  const int nsub = (wv >> 2) * 32;

  if (tid < 64) {
    const int   o  = o0 + tid;
    const float iv = g[o] * rsqrtf(vv[o] + BN_EPS_);
    inv_s[tid]  = iv;
    beta_s[tid] = bb[o] - mm[o] * iv;
  }

  v8f acc[2] = {{}, {}};
  for (int ct = 0; ct < C_ / 32; ++ct) {
    __syncthreads();
    {  // Ws: 64 rows x 32 k, vectorized f32->bf16, contiguous writes
      const int r = tid >> 2, s = (tid & 3) * 8;
      const float* src = w + (size_t)(o0 + r) * C_ + ct * 32 + s;
      v8bf t;
#pragma unroll
      for (int j = 0; j < 8; ++j) t[j] = (__bf16)src[j];
      *(v8bf*)&Ws[r * LDA_ + s] = t;
    }
    {  // Xs transposed [n][c]: coalesced f32 reads along n, b16 scatter into LDS
      const int c = tid >> 3, ns = (tid & 7) * 8;
      const float* src = x + ((size_t)b * C_ + ct * 32 + c) * N_ + n0 + ns;
#pragma unroll
      for (int j = 0; j < 8; ++j) Xs[(ns + j) * LDA_ + c] = (__bf16)src[j];
    }
    __syncthreads();
    const v16bf a = load_a_frag(&Ws[osub * LDA_], lane);
    acc[0] = wmma_bf16(a, load_b_frag(Xs, lane, nsub), acc[0]);
    acc[1] = wmma_bf16(a, load_b_frag(Xs, lane, nsub + 16), acc[1]);
  }

  const int hi   = (lane >> 4) & 1;
  const int oT   = o0 + osub;          // tile base channel (uniform per wave)
  const int head = oT >> 7;
  const int cp0  = oT & 127;           // region: [0,32)=q [32,64)=k [64,128)=v
  const int bh   = b * NH_ + head;

#pragma unroll
  for (int t = 0; t < 2; ++t) {
    const int n = n0 + nsub + t * 16 + (lane & 15);
    v8bf  ov;
    float fv[8];
#pragma unroll
    for (int v = 0; v < 8; ++v) {
      const int rloc = osub + v + hi * 8;
      fv[v] = acc[t][v] * inv_s[rloc] + beta_s[rloc];
      ov[v] = (__bf16)fv[v];
    }
    if (cp0 < DK_) {
      *(v8bf*)&qT[((size_t)bh * N_ + n) * DK_ + cp0 + hi * 8] = ov;
    } else if (cp0 < 2 * DK_) {
      *(v8bf*)&kT[((size_t)bh * N_ + n) * DK_ + (cp0 - DK_) + hi * 8] = ov;
    } else {
      const int d0 = cp0 - 2 * DK_ + hi * 8;
#pragma unroll
      for (int v = 0; v < 8; ++v) {
        vW[((size_t)bh * DV_ + d0 + v) * N_ + n]              = ov[v];
        vimg[((size_t)b * C_ + head * DV_ + d0 + v) * N_ + n] = fv[v];
      }
    }
  }
}

// ---------------------------------------------------------------------------
// Kernel 2: two-pass flash attention. One block = one (b,h) x 128 query rows.
// ---------------------------------------------------------------------------
__global__ __launch_bounds__(256) void attn_kernel(
    const __bf16* __restrict__ qT, const __bf16* __restrict__ kT,
    const __bf16* __restrict__ vW, float* __restrict__ attnout) {
  __shared__ __align__(16) char smem[64 * 132 * 4];  // 33792 B, two overlays
  __bf16* Qs = (__bf16*)smem;            // [128][LDA_] q rows (A source)
  __bf16* Ks = Qs + 128 * LDA_;          // [32 keys][LDA_] (B source, [n][k])
  __bf16* Vs = Ks + 32 * LDA_;           // [64 d][LDA_]   (B source, [n][k])
  __bf16* Ps = Vs + 64 * LDA_;           // [8 waves][16 q][LDA_] P tiles
  float*  Ot = (float*)smem;             // epilogue overlay: [64 d][132 q]

  const int bh   = blockIdx.x;
  const int b    = bh / NH_;
  const int h    = bh % NH_;
  const int q0   = blockIdx.y * 128;
  const int tid  = threadIdx.x;
  const int lane = tid & 31;
  const int wv   = tid >> 5;
  const int hi   = (lane >> 4) & 1;

  for (int i = tid; i < 128 * 4; i += 256) {
    const int r = i >> 2, s = (i & 3) * 8;
    *(v8bf*)&Qs[r * LDA_ + s] = *(const v8bf*)&qT[((size_t)bh * N_ + q0 + r) * DK_ + s];
  }
  __syncthreads();
  const v16bf aq = load_a_frag(&Qs[wv * 16 * LDA_], lane);

  float mrun[8], lrun[8];
#pragma unroll
  for (int v = 0; v < 8; ++v) { mrun[v] = -3.0e38f; lrun[v] = 0.0f; }

  const int NKT = N_ / 32;

  // ---- Pass 1: online softmax statistics ----
  for (int kt = 0; kt < NKT; ++kt) {
    __syncthreads();
    if (tid < 128) {
      const int r = tid >> 2, s = (tid & 3) * 8;
      *(v8bf*)&Ks[r * LDA_ + s] =
          *(const v8bf*)&kT[((size_t)bh * N_ + kt * 32 + r) * DK_ + s];
    }
    if (kt + 1 < NKT)
      __builtin_prefetch(&kT[((size_t)bh * N_ + (kt + 1) * 32) * DK_ + tid * 4], 0, 3);
    __syncthreads();
    v8f z = {};
    const v8f s0 = wmma_bf16(aq, load_b_frag(Ks, lane, 0), z);
    const v8f s1 = wmma_bf16(aq, load_b_frag(Ks, lane, 16), z);
#pragma unroll
    for (int v = 0; v < 8; ++v) {
      const float x0   = s0[v] * SCALE_;
      const float x1   = s1[v] * SCALE_;
      const float rmax = red_max16(fmaxf(x0, x1));
      const float nm   = fmaxf(mrun[v], rmax);
      const float ts   = red_sum16(__expf(x0 - nm) + __expf(x1 - nm));
      lrun[v] = lrun[v] * __expf(mrun[v] - nm) + ts;
      mrun[v] = nm;
    }
  }

  // ---- Pass 2: O = sum P * V ----
  v8f oa0 = {}, oa1 = {}, oa2 = {}, oa3 = {};
  __bf16* Pw = Ps + wv * 16 * LDA_;
  for (int kt = 0; kt < NKT; ++kt) {
    __syncthreads();
    if (tid < 128) {
      const int r = tid >> 2, s = (tid & 3) * 8;
      *(v8bf*)&Ks[r * LDA_ + s] =
          *(const v8bf*)&kT[((size_t)bh * N_ + kt * 32 + r) * DK_ + s];
    }
    {
      const int d = tid >> 2, s = (tid & 3) * 8;
      *(v8bf*)&Vs[d * LDA_ + s] =
          *(const v8bf*)&vW[((size_t)bh * DV_ + d) * N_ + kt * 32 + s];
    }
    if (kt + 1 < NKT)
      __builtin_prefetch(&vW[(size_t)bh * DV_ * N_ + (kt + 1) * 32 + (tid >> 2) * N_], 0, 3);
    __syncthreads();
    v8f z = {};
    const v8f s0 = wmma_bf16(aq, load_b_frag(Ks, lane, 0), z);
    const v8f s1 = wmma_bf16(aq, load_b_frag(Ks, lane, 16), z);
#pragma unroll
    for (int v = 0; v < 8; ++v) {
      const float p0 = __expf(s0[v] * SCALE_ - mrun[v]);
      const float p1 = __expf(s1[v] * SCALE_ - mrun[v]);
      Pw[(v + hi * 8) * LDA_ + (lane & 15)]      = (__bf16)p0;  // C-layout -> [q][key]
      Pw[(v + hi * 8) * LDA_ + 16 + (lane & 15)] = (__bf16)p1;
    }
    __syncthreads();
    const v16bf ap = load_a_frag(Pw, lane);
    oa0 = wmma_bf16(ap, load_b_frag(Vs, lane, 0),  oa0);
    oa1 = wmma_bf16(ap, load_b_frag(Vs, lane, 16), oa1);
    oa2 = wmma_bf16(ap, load_b_frag(Vs, lane, 32), oa2);
    oa3 = wmma_bf16(ap, load_b_frag(Vs, lane, 48), oa3);
  }

  // ---- Epilogue: transpose O through LDS, store coalesced ----
  __syncthreads();
#pragma unroll
  for (int v = 0; v < 8; ++v) {
    const float rcp = 1.0f / lrun[v];
    const int   q   = wv * 16 + v + hi * 8;   // 0..127
    const int   dl  = lane & 15;
    Ot[(0  + dl) * 132 + q] = oa0[v] * rcp;
    Ot[(16 + dl) * 132 + q] = oa1[v] * rcp;
    Ot[(32 + dl) * 132 + q] = oa2[v] * rcp;
    Ot[(48 + dl) * 132 + q] = oa3[v] * rcp;
  }
  __syncthreads();
  for (int i = tid; i < 64 * 128; i += 256) {
    const int ch = i >> 7, q = i & 127;
    attnout[((size_t)b * C_ + h * DV_ + ch) * N_ + q0 + q] = Ot[ch * 132 + q];
  }
}

// ---------------------------------------------------------------------------
// Kernel 3: depthwise 3x3 conv on V image + BN, added into attention output.
// ---------------------------------------------------------------------------
__global__ __launch_bounds__(256) void pe_kernel(
    const float* __restrict__ vimg, const float* __restrict__ pw,
    const float* __restrict__ pg, const float* __restrict__ pb,
    const float* __restrict__ pm, const float* __restrict__ pv,
    float* __restrict__ attnout) {
  const int bc = blockIdx.x;
  const int c  = bc % C_;
  const int n  = blockIdx.y * 256 + threadIdx.x;
  const int y  = n / HGRID_;
  const int xx = n % HGRID_;
  float s = 0.0f;
#pragma unroll
  for (int ky = 0; ky < 3; ++ky) {
#pragma unroll
    for (int kx = 0; kx < 3; ++kx) {
      const int yy = y + ky - 1;
      const int xw = xx + kx - 1;
      if (yy >= 0 && yy < HGRID_ && xw >= 0 && xw < HGRID_)
        s += pw[c * 9 + ky * 3 + kx] * vimg[(size_t)bc * N_ + yy * HGRID_ + xw];
    }
  }
  const float inv  = pg[c] * rsqrtf(pv[c] + BN_EPS_);
  const float beta = pb[c] - pm[c] * inv;
  attnout[(size_t)bc * N_ + n] += s * inv + beta;
}

// ---------------------------------------------------------------------------
// Kernel 4: Proj GEMM (bf16 WMMA) + BN -> final f32 NCHW output.
// ---------------------------------------------------------------------------
__global__ __launch_bounds__(256) void proj_gemm_kernel(
    const float* __restrict__ ain, const float* __restrict__ w,
    const float* __restrict__ g, const float* __restrict__ bb,
    const float* __restrict__ mm, const float* __restrict__ vv,
    float* __restrict__ out) {
  __shared__ __align__(16) __bf16 Ws[64 * LDA_];
  __shared__ __align__(16) __bf16 Xs[64 * LDA_];
  __shared__ float inv_s[64];
  __shared__ float beta_s[64];

  const int b    = blockIdx.x;
  const int o0   = blockIdx.y * 64;
  const int n0   = blockIdx.z * 64;
  const int tid  = threadIdx.x;
  const int lane = tid & 31;
  const int wv   = tid >> 5;
  const int osub = (wv & 3) * 16;
  const int nsub = (wv >> 2) * 32;

  if (tid < 64) {
    const int   o  = o0 + tid;
    const float iv = g[o] * rsqrtf(vv[o] + BN_EPS_);
    inv_s[tid]  = iv;
    beta_s[tid] = bb[o] - mm[o] * iv;
  }

  v8f acc[2] = {{}, {}};
  for (int ct = 0; ct < C_ / 32; ++ct) {
    __syncthreads();
    {
      const int r = tid >> 2, s = (tid & 3) * 8;
      const float* src = w + (size_t)(o0 + r) * C_ + ct * 32 + s;
      v8bf t;
#pragma unroll
      for (int j = 0; j < 8; ++j) t[j] = (__bf16)src[j];
      *(v8bf*)&Ws[r * LDA_ + s] = t;
    }
    {
      const int c = tid >> 3, ns = (tid & 7) * 8;
      const float* src = ain + ((size_t)b * C_ + ct * 32 + c) * N_ + n0 + ns;
#pragma unroll
      for (int j = 0; j < 8; ++j) Xs[(ns + j) * LDA_ + c] = (__bf16)src[j];
    }
    __syncthreads();
    const v16bf a = load_a_frag(&Ws[osub * LDA_], lane);
    acc[0] = wmma_bf16(a, load_b_frag(Xs, lane, nsub), acc[0]);
    acc[1] = wmma_bf16(a, load_b_frag(Xs, lane, nsub + 16), acc[1]);
  }

  const int hi = (lane >> 4) & 1;
#pragma unroll
  for (int t = 0; t < 2; ++t) {
    const int n = n0 + nsub + t * 16 + (lane & 15);
#pragma unroll
    for (int v = 0; v < 8; ++v) {
      const int rloc = osub + v + hi * 8;
      out[((size_t)b * C_ + o0 + rloc) * N_ + n] = acc[t][v] * inv_s[rloc] + beta_s[rloc];
    }
  }
}

// ---------------------------------------------------------------------------
extern "C" void kernel_launch(void* const* d_in, const int* in_sizes, int n_in,
                              void* d_out, int out_size, void* d_ws, size_t ws_size,
                              hipStream_t stream) {
  const float* x      = (const float*)d_in[0];
  const float* qkv_w  = (const float*)d_in[1];
  const float* qkv_g  = (const float*)d_in[2];
  const float* qkv_b  = (const float*)d_in[3];
  const float* qkv_m  = (const float*)d_in[4];
  const float* qkv_v  = (const float*)d_in[5];
  const float* pe_w   = (const float*)d_in[6];
  const float* pe_g   = (const float*)d_in[7];
  const float* pe_b   = (const float*)d_in[8];
  const float* pe_m   = (const float*)d_in[9];
  const float* pe_v   = (const float*)d_in[10];
  const float* proj_w = (const float*)d_in[11];
  const float* proj_g = (const float*)d_in[12];
  const float* proj_b = (const float*)d_in[13];
  const float* proj_m = (const float*)d_in[14];
  const float* proj_v = (const float*)d_in[15];

  char* wsp = (char*)d_ws;
  __bf16* qT = (__bf16*)wsp;  wsp += (size_t)B_ * NH_ * N_ * DK_ * sizeof(__bf16);
  __bf16* kT = (__bf16*)wsp;  wsp += (size_t)B_ * NH_ * N_ * DK_ * sizeof(__bf16);
  __bf16* vW = (__bf16*)wsp;  wsp += (size_t)B_ * NH_ * DV_ * N_ * sizeof(__bf16);
  float* vimg = (float*)wsp;  wsp += (size_t)B_ * C_ * N_ * sizeof(float);
  float* aout = (float*)wsp;  wsp += (size_t)B_ * C_ * N_ * sizeof(float);

  qkv_gemm_kernel<<<dim3(B_, HQKV_ / 64, N_ / 64), 256, 0, stream>>>(
      x, qkv_w, qkv_g, qkv_b, qkv_m, qkv_v, qT, kT, vW, vimg);

  attn_kernel<<<dim3(B_ * NH_, N_ / 128), 256, 0, stream>>>(qT, kT, vW, aout);

  pe_kernel<<<dim3(B_ * C_, N_ / 256), 256, 0, stream>>>(
      vimg, pe_g ? pe_w : pe_w, pe_g, pe_b, pe_m, pe_v, aout);

  proj_gemm_kernel<<<dim3(B_, C_ / 64, N_ / 64), 256, 0, stream>>>(
      aout, proj_w, proj_g, proj_b, proj_m, proj_v, (float*)d_out);
}
